// DLRM_48172353192217
// MI455X (gfx1250) — matrix operand order repaired and checked by reading, not compiled, not run on hardware
//
#include <hip/hip_runtime.h>
#include <hip/hip_bf16.h>

// ---------------------------------------------------------------------------
// DLRM forward for MI455X (gfx1250, wave32, WMMA).
// All GEMMs run on v_wmma_f32_16x16x32_f16 (f16 in, f32 accumulate).
// GEMM: 256x64 workgroup tile, 32x64 wave tile, LDS-double-buffered W tiles,
// one barrier per K-step, 8 WMMAs per wave per K-step.
// ---------------------------------------------------------------------------

typedef __attribute__((ext_vector_type(16))) _Float16 v16h;
typedef __attribute__((ext_vector_type(8)))  _Float16 v8h;
typedef __attribute__((ext_vector_type(4)))  _Float16 v4h;
typedef __attribute__((ext_vector_type(8)))  float    v8f;

#define BATCH     8192
#define NUM_DENSE 13
#define NUM_CAT   26
#define XCOLS     39      // 13 dense + 26 cat
#define VOCAB     100000
#define EMBED     128
#define TOPIN_PAD 512     // 128 + 378 = 506, padded to 512

// ---------------------------------------------------------------------------
// Pack dense features [B,13] fp32 -> [B,32] f16 (zero padded K)
// ---------------------------------------------------------------------------
__global__ __launch_bounds__(256) void pack_dense_f16(
    const float* __restrict__ x, _Float16* __restrict__ A0) {
  int t = blockIdx.x * 256 + threadIdx.x;         // BATCH*32 threads
  int b = t >> 5, j = t & 31;
  float v = (j < NUM_DENSE) ? x[(size_t)b * XCOLS + j] : 0.f;
  A0[t] = (_Float16)v;
}

// ---------------------------------------------------------------------------
// Convert a [K,N] fp32 weight to [Kpad,N] f16 (zero rows K..Kpad)
// ---------------------------------------------------------------------------
__global__ __launch_bounds__(256) void convert_weight_f16(
    const float* __restrict__ src, _Float16* __restrict__ dst,
    int K, int N, int Kpad) {
  int t = blockIdx.x * 256 + threadIdx.x;
  if (t >= Kpad * N) return;
  int k = t / N;
  dst[t] = (k < K) ? (_Float16)src[t] : (_Float16)0.f;
}

// ---------------------------------------------------------------------------
// WMMA GEMM: Out[M,N] (f16) = relu(A[M,K] (f16) @ W[K,N] (f16) + bias (f32))
// K multiple of 32, N multiple of 64, M multiple of 256.
// Block = 256 threads (8 waves); workgroup tile 256(M) x 64(N);
// wave tile 32(M) x 64(N). W tile double-buffered, transposed, in LDS.
// ---------------------------------------------------------------------------
__global__ __launch_bounds__(256) void wmma_gemm_bias_relu_f16(
    const _Float16* __restrict__ A, const _Float16* __restrict__ W,
    const float* __restrict__ bias, _Float16* __restrict__ Out,
    int N, int K) {
  constexpr int LDT = 40;                       // halves; 80B rows -> 16B aligned
  __shared__ __align__(16) _Float16 wt[2][64 * LDT];  // double-buffered W tile

  const int tid  = threadIdx.x;
  const int wave = tid >> 5;
  const int lane = tid & 31;
  const int cb   = blockIdx.x * 64;             // column base of block
  const int mrow = blockIdx.y * 256 + wave * 32;// this wave's 32-row stripe

  const int nloc = lane & 15;
  const int kh   = (lane >> 4) & 1;             // lane half select

  // staging: global row-major W -> transposed LDS [n][k]
  const int sr = tid >> 3;                      // k within tile (0..31)
  const int sc = (tid & 7) * 8;                 // n within tile (0..56)
  auto stage = [&](int buf, int kk) {
    const v8h wv = *(const v8h*)(W + (size_t)(kk + sr) * N + cb + sc);
#pragma unroll
    for (int i = 0; i < 8; ++i) wt[buf][(sc + i) * LDT + sr] = wv[i];
  };

  v8f acc[2][4];
#pragma unroll
  for (int t = 0; t < 2; ++t)
#pragma unroll
    for (int j = 0; j < 4; ++j) acc[t][j] = v8f{};

  stage(0, 0);
  __syncthreads();

  int p = 0;
  for (int kk = 0; kk < K; kk += 32) {
    if (kk + 32 < K) {
      stage(p ^ 1, kk + 32);                    // overlap: fill other buffer
      if (kk + 64 < K)                          // hint tile after that
        __builtin_prefetch(W + (size_t)(kk + 64) * N + cb, 0, 1);
    }

    // ---- two A fragments 16x32 from global (documented 16-bit A layout) ----
    // lanes 0-15: halves K 0-7 / 16-23 ; lanes 16-31: K 8-15 / 24-31
    v16h a[2];
#pragma unroll
    for (int t = 0; t < 2; ++t) {
      const _Float16* arow =
          A + (size_t)(mrow + t * 16 + nloc) * K + kk + kh * 8;
      const v8h alo = *(const v8h*)(arow);
      const v8h ahi = *(const v8h*)(arow + 16);
#pragma unroll
      for (int i = 0; i < 8; ++i) { a[t][i] = alo[i]; a[t][8 + i] = ahi[i]; }
    }

    // ---- 4 B fragments (32x16) from transposed LDS; 8 WMMAs ----
#pragma unroll
    for (int j = 0; j < 4; ++j) {
      const _Float16* bcol = &wt[p][(j * 16 + nloc) * LDT + kh * 8];
      const v8h blo = *(const v8h*)(bcol);
      const v8h bhi = *(const v8h*)(bcol + 16);
      v16h bf;
#pragma unroll
      for (int i = 0; i < 8; ++i) { bf[i] = blo[i]; bf[8 + i] = bhi[i]; }
      acc[0][j] = __builtin_amdgcn_wmma_f32_16x16x32_f16(
          false, a[0], false, bf, (short)0, acc[0][j], false, false);
      acc[1][j] = __builtin_amdgcn_wmma_f32_16x16x32_f16(
          false, a[1], false, bf, (short)0, acc[1][j], false, false);
    }
    __syncthreads();                            // one barrier per K-step
    p ^= 1;
  }

  // ---- epilogue: bias + relu + f16 store (lanes 0-15 coalesce per row) ----
#pragma unroll
  for (int t = 0; t < 2; ++t) {
    const int mb = mrow + t * 16 + kh * 8;      // C layout: lanes16-31 -> M+8
#pragma unroll
    for (int j = 0; j < 4; ++j) {
      const int col = cb + j * 16 + nloc;
      const float bv = bias[col];
#pragma unroll
      for (int v = 0; v < 8; ++v) {
        float r = acc[t][j][v] + bv;
        r = r > 0.f ? r : 0.f;
        Out[(size_t)(mb + v) * N + col] = (_Float16)r;
      }
    }
  }
}

// ---------------------------------------------------------------------------
// Fused: embedding gather + 27x27 interaction Gram (WMMA) + top-input build.
// One wave per sample; block = 128 threads (4 samples).
// F = [32,128] f16 tile in LDS: row 0 = h, rows 1..26 = embeddings,
// rows 27..31 = 0. G = F @ F^T, upper triangle -> topin[:,128:506].
// topin[:,0:128] = h, topin[:,506:512] = 0.
// ---------------------------------------------------------------------------
__global__ __launch_bounds__(128) void interact_kernel(
    const float* __restrict__ x, const float* __restrict__ emb,
    const _Float16* __restrict__ h, _Float16* __restrict__ topin) {
  constexpr int LDF = 136;                      // halves; 272B rows -> aligned
  __shared__ __align__(16) _Float16 F[4 * 32 * LDF];

  const int wave = threadIdx.x >> 5;
  const int lane = threadIdx.x & 31;
  const int b    = blockIdx.x * 4 + wave;
  _Float16* Fw   = &F[wave * 32 * LDF];
  const float* xrow = x + (size_t)b * XCOLS;
  _Float16* trow = topin + (size_t)b * TOPIN_PAD;

  // feature 0 = bottom MLP output; also copy straight into topin[0:128]
  {
    const v4h hv = *(const v4h*)(h + (size_t)b * EMBED + lane * 4);
    *(v4h*)(&Fw[lane * 4]) = hv;
    *(v4h*)(&trow[lane * 4]) = hv;
  }
  // features 1..26 = embedding rows (single pass over the HBM table)
  for (int c = 0; c < NUM_CAT; ++c) {
    const int gidx = (int)xrow[NUM_DENSE + c] + c * VOCAB;
    const float4 ev = *(const float4*)(emb + (size_t)gidx * EMBED + lane * 4);
    v4h evh;
    evh[0] = (_Float16)ev.x; evh[1] = (_Float16)ev.y;
    evh[2] = (_Float16)ev.z; evh[3] = (_Float16)ev.w;
    *(v4h*)(&Fw[(1 + c) * LDF + lane * 4]) = evh;
  }
  // zero pad rows 27..31
  {
    v4h z = {};
#pragma unroll
    for (int r = 27; r < 32; ++r) *(v4h*)(&Fw[r * LDF + lane * 4]) = z;
  }
  if (lane < 6) trow[506 + lane] = (_Float16)0.f;  // zero K-pad of top input
  __syncthreads();

  // Gram = F(32x128) @ F^T(128x32) : 2x2 tiles x 4 K-steps = 16 WMMAs
  const int nloc = lane & 15;
  const int kh   = (lane >> 4) & 1;
#pragma unroll
  for (int mt = 0; mt < 2; ++mt) {
#pragma unroll
    for (int nt = 0; nt < 2; ++nt) {
      v8f acc = {};
#pragma unroll
      for (int kk = 0; kk < EMBED; kk += 32) {
        const _Float16* ar = &Fw[(mt * 16 + nloc) * LDF + kk + kh * 8];
        const v8h alo = *(const v8h*)(ar);
        const v8h ahi = *(const v8h*)(ar + 16);
        v16h a;
#pragma unroll
        for (int i = 0; i < 8; ++i) { a[i] = alo[i]; a[8 + i] = ahi[i]; }
        // B[k][n] = F[nt*16+n][k]  (transpose comes free from layout)
        const _Float16* br = &Fw[(nt * 16 + nloc) * LDF + kk + kh * 8];
        const v8h blo = *(const v8h*)(br);
        const v8h bhi = *(const v8h*)(br + 16);
        v16h bf;
#pragma unroll
        for (int i = 0; i < 8; ++i) { bf[i] = blo[i]; bf[8 + i] = bhi[i]; }
        acc = __builtin_amdgcn_wmma_f32_16x16x32_f16(
            false, a, false, bf, (short)0, acc, false, false);
      }
      // scatter upper triangle (incl. diagonal) into topin[:,128:506]
      const int n  = nt * 16 + nloc;
      const int mb = mt * 16 + kh * 8;
#pragma unroll
      for (int v = 0; v < 8; ++v) {
        const int m = mb + v;
        if (m <= n && n < 27) {
          const int off = m * 27 - (m * (m - 1)) / 2 + (n - m);
          trow[EMBED + off] = (_Float16)acc[v];
        }
      }
    }
  }
}

// ---------------------------------------------------------------------------
// Final 256 -> 1 layer (tiny): one wave per row, fp32 output, no relu.
// ---------------------------------------------------------------------------
__global__ __launch_bounds__(256) void final_layer_kernel(
    const _Float16* __restrict__ t4, const float* __restrict__ w,
    const float* __restrict__ bias, float* __restrict__ out) {
  const int b    = blockIdx.x * 8 + (threadIdx.x >> 5);
  const int lane = threadIdx.x & 31;
  float s = 0.f;
#pragma unroll
  for (int k = lane; k < 256; k += 32)
    s += (float)t4[(size_t)b * 256 + k] * w[k];
#pragma unroll
  for (int off = 16; off > 0; off >>= 1) s += __shfl_xor(s, off, 32);
  if (lane == 0) out[b] = s + bias[0];
}

// ---------------------------------------------------------------------------
// Host orchestration
// ---------------------------------------------------------------------------
extern "C" void kernel_launch(void* const* d_in, const int* in_sizes, int n_in,
                              void* d_out, int out_size, void* d_ws, size_t ws_size,
                              hipStream_t stream) {
  (void)in_sizes; (void)n_in; (void)out_size; (void)ws_size;
  const float* x   = (const float*)d_in[0];
  const float* emb = (const float*)d_in[1];
  const float* bw[3] = {(const float*)d_in[2], (const float*)d_in[3], (const float*)d_in[4]};
  const float* bb[3] = {(const float*)d_in[5], (const float*)d_in[6], (const float*)d_in[7]};
  const float* tw[5] = {(const float*)d_in[8], (const float*)d_in[9], (const float*)d_in[10],
                        (const float*)d_in[11], (const float*)d_in[12]};
  const float* tb[5] = {(const float*)d_in[13], (const float*)d_in[14], (const float*)d_in[15],
                        (const float*)d_in[16], (const float*)d_in[17]};
  float* out = (float*)d_out;

  char* ws = (char*)d_ws;
  size_t off = 0;
  auto alloc = [&](size_t bytes) -> void* {
    void* p = ws + off;
    off = (off + bytes + 255) & ~(size_t)255;
    return p;
  };

  _Float16* A0    = (_Float16*)alloc((size_t)BATCH * 32 * 2);
  _Float16* act1  = (_Float16*)alloc((size_t)BATCH * 512 * 2);
  _Float16* act2  = (_Float16*)alloc((size_t)BATCH * 256 * 2);
  _Float16* hbuf  = (_Float16*)alloc((size_t)BATCH * 128 * 2);
  _Float16* topin = (_Float16*)alloc((size_t)BATCH * TOPIN_PAD * 2);
  _Float16* t1    = (_Float16*)alloc((size_t)BATCH * 1024 * 2);
  _Float16* t2    = (_Float16*)alloc((size_t)BATCH * 1024 * 2);
  _Float16* t3    = (_Float16*)alloc((size_t)BATCH * 512 * 2);
  _Float16* t4    = (_Float16*)alloc((size_t)BATCH * 256 * 2);
  _Float16* Wb0   = (_Float16*)alloc((size_t)32 * 512 * 2);
  _Float16* Wb1   = (_Float16*)alloc((size_t)512 * 256 * 2);
  _Float16* Wb2   = (_Float16*)alloc((size_t)256 * 128 * 2);
  _Float16* Wt0   = (_Float16*)alloc((size_t)512 * 1024 * 2);
  _Float16* Wt1   = (_Float16*)alloc((size_t)1024 * 1024 * 2);
  _Float16* Wt2   = (_Float16*)alloc((size_t)1024 * 512 * 2);
  _Float16* Wt3   = (_Float16*)alloc((size_t)512 * 256 * 2);

  // --- input / weight conversion (f32 -> f16, zero-padded K) ---
  pack_dense_f16<<<(BATCH * 32) / 256, 256, 0, stream>>>(x, A0);
  auto conv = [&](const float* s, _Float16* d, int K, int N, int Kp) {
    int tot = Kp * N;
    convert_weight_f16<<<(tot + 255) / 256, 256, 0, stream>>>(s, d, K, N, Kp);
  };
  conv(bw[0], Wb0, 13, 512, 32);
  conv(bw[1], Wb1, 512, 256, 512);
  conv(bw[2], Wb2, 256, 128, 256);
  conv(tw[0], Wt0, 506, 1024, 512);
  conv(tw[1], Wt1, 1024, 1024, 1024);
  conv(tw[2], Wt2, 1024, 512, 1024);
  conv(tw[3], Wt3, 512, 256, 512);

  auto gemm = [&](const _Float16* Ai, const _Float16* Wi, const float* bi,
                  _Float16* Oi, int N, int K) {
    dim3 g(N / 64, BATCH / 256);
    wmma_gemm_bias_relu_f16<<<g, 256, 0, stream>>>(Ai, Wi, bi, Oi, N, K);
  };

  // --- bottom MLP ---
  gemm(A0, Wb0, bb[0], act1, 512, 32);
  gemm(act1, Wb1, bb[1], act2, 256, 512);
  gemm(act2, Wb2, bb[2], hbuf, 128, 256);

  // --- fused embedding gather + interaction + top-input assembly ---
  interact_kernel<<<BATCH / 4, 128, 0, stream>>>(x, emb, hbuf, topin);

  // --- top MLP ---
  gemm(topin, Wt0, tb[0], t1, 1024, 512);
  gemm(t1, Wt1, tb[1], t2, 1024, 1024);
  gemm(t2, Wt2, tb[2], t3, 512, 1024);
  gemm(t3, Wt3, tb[3], t4, 256, 512);

  // --- final 256 -> 1 layer, fp32 output ---
  final_layer_kernel<<<BATCH / 8, 256, 0, stream>>>(t4, tw[4], tb[4], out);
}
